// RNN_79293686219040
// MI455X (gfx1250) — matrix operand order repaired and checked by reading, not compiled, not run on hardware
//
#include <hip/hip_runtime.h>
#include <cstdint>

// ---------------- problem constants ----------------
#define BB 1024
#define TT 512
#define FF 64
#define HH 128
#define GG 512            // 4*H gate width

typedef __attribute__((ext_vector_type(16))) __bf16 v16bf;
typedef __attribute__((ext_vector_type(8)))  float  v8f;

struct Frag32 { uint4 lo; uint4 hi; };   // 32 bytes == v16bf

__device__ __forceinline__ unsigned short f2bf(float f) {
    uint32_t u = __builtin_bit_cast(uint32_t, f);
    u += 0x7fffu + ((u >> 16) & 1u);     // round-to-nearest-even
    return (unsigned short)(u >> 16);
}
__device__ __forceinline__ float sigmoidf(float x) {
    return 1.f / (1.f + __expf(-x));
}
__device__ __forceinline__ float tanh_fast(float x) {
    x = fminf(fmaxf(x, -15.f), 15.f);
    float e = __expf(2.f * x);
    return (e - 1.f) / (e + 1.f);
}

// ---------------- weight/bias packing ----------------
// Wcat0: [512][192] bf16 = [W_ih0 | W_hh0], Wcat1: [512][256] bf16 = [W_ih1 | W_hh1]
__global__ void prep_kernel(const float* __restrict__ Wih0, const float* __restrict__ Whh0,
                            const float* __restrict__ bih0, const float* __restrict__ bhh0,
                            const float* __restrict__ Wih1, const float* __restrict__ Whh1,
                            const float* __restrict__ bih1, const float* __restrict__ bhh1,
                            unsigned short* __restrict__ Wcat0, unsigned short* __restrict__ Wcat1,
                            float* __restrict__ bias0, float* __restrict__ bias1) {
    int i = blockIdx.x * blockDim.x + threadIdx.x;
    if (i < GG * 192) {
        int g = i / 192, k = i % 192;
        float v = (k < FF) ? Wih0[g * FF + k] : Whh0[g * HH + (k - FF)];
        Wcat0[i] = f2bf(v);
    }
    if (i < GG * 256) {
        int g = i / 256, k = i % 256;
        float v = (k < HH) ? Wih1[g * HH + k] : Whh1[g * HH + (k - HH)];
        Wcat1[i] = f2bf(v);
    }
    if (i < GG) {
        bias0[i] = bih0[i] + bhh0[i];
        bias1[i] = bih1[i] + bhh1[i];
    }
}

// ---------------- fused LSTM layer (persistent over T) ----------------
// Block: 512 threads = 16 waves, owns 16 batch rows for all 512 timesteps.
// Wave w owns gate columns [32w, 32w+32): 2 N-tiles, full K in VGPR-resident B-frags.
template<int KTOT, bool FIRST>
__global__ __launch_bounds__(512, 1)
void lstm_pass(const float* __restrict__ x,            // pass A input [B,T,64] f32
               const unsigned short* __restrict__ hin, // pass B input [B,T,128] bf16
               const unsigned short* __restrict__ Wcat,// [512][KTOT] bf16
               const float* __restrict__ bias,         // [512] f32 (b_ih+b_hh)
               unsigned short* __restrict__ hout,      // pass A out [B,T,128] bf16
               float* __restrict__ h2last)             // pass B out [B,128] f32
{
    constexpr int KB   = KTOT / 32;          // k-blocks of 32
    constexpr int KOFF = FIRST ? FF : HH;    // where h lives inside [input|h]

    __shared__ unsigned short Abuf[16 * KTOT];   // [x_t | h] bf16, 16 rows
    __shared__ float gatesLds[16 * GG];          // f32 gate pre-activations
    __shared__ float biasLds[GG];

    const int tid  = threadIdx.x;
    const int w    = tid >> 5;
    const int lane = tid & 31;
    const int n    = lane & 15;
    const int half = lane >> 4;
    const int bbase = blockIdx.x * 16;

    biasLds[tid] = bias[tid];                            // tid in [0,512)
    for (int i = tid; i < 16 * KTOT; i += 512) Abuf[i] = 0;

    // ---- preload B fragments into VGPRs (kept live across all timesteps) ----
    v16bf bw[2][KB];
#pragma unroll
    for (int nt = 0; nt < 2; ++nt) {
#pragma unroll
        for (int kb = 0; kb < KB; ++kb) {
            const unsigned short* p =
                Wcat + (size_t)(32 * w + nt * 16 + n) * KTOT + kb * 32 + 16 * half;
            Frag32 f;
            f.lo = *(const uint4*)p;          // K = 16*half + 0..7
            f.hi = *(const uint4*)(p + 8);    // K = 16*half + 8..15
            bw[nt][kb] = __builtin_bit_cast(v16bf, f);
        }
    }

    float cst[4] = {0.f, 0.f, 0.f, 0.f};     // per-thread cell-state slice
    __syncthreads();

    for (int t = 0; t < TT; ++t) {
        // ---- stage this timestep's input into Abuf (bf16) ----
        if constexpr (FIRST) {
#pragma unroll
            for (int r = 0; r < 2; ++r) {
                int idx = tid + 512 * r;
                int row = idx >> 6, f = idx & 63;
                const float* src = &x[((size_t)(bbase + row) * TT + t) * FF + f];
                Abuf[row * KTOT + f] = f2bf(*src);
                if (t + 1 < TT) __builtin_prefetch(src + FF, 0, 3);
            }
        } else {
#pragma unroll
            for (int r = 0; r < 4; ++r) {
                int idx = tid + 512 * r;
                int row = idx >> 7, f = idx & 127;
                const unsigned short* src = &hin[((size_t)(bbase + row) * TT + t) * HH + f];
                Abuf[row * KTOT + f] = *src;
                if (t + 1 < TT) __builtin_prefetch(src + HH, 0, 3);
            }
        }
        __syncthreads();   // input + previous h visible to all waves

        // ---- gate GEMM: [16,KTOT] x [KTOT,32] per wave via WMMA bf16 ----
        v8f acc0 = {0.f,0.f,0.f,0.f,0.f,0.f,0.f,0.f};
        v8f acc1 = {0.f,0.f,0.f,0.f,0.f,0.f,0.f,0.f};
#pragma unroll
        for (int kb = 0; kb < KB; ++kb) {
            // A-frag: contiguous k order, lane half h holds K = 16*h + 0..15 (matches B)
            const unsigned short* pa = &Abuf[n * KTOT + kb * 32 + 16 * half];
            Frag32 fa;
            fa.lo = *(const uint4*)pa;          // K = kb*32 + 16*half + 0..7
            fa.hi = *(const uint4*)(pa + 8);    // K = kb*32 + 16*half + 8..15
            v16bf av = __builtin_bit_cast(v16bf, fa);
            acc0 = __builtin_amdgcn_wmma_f32_16x16x32_bf16(
                false, av, false, bw[0][kb], (short)0, acc0, false, false);
            acc1 = __builtin_amdgcn_wmma_f32_16x16x32_bf16(
                false, av, false, bw[1][kb], (short)0, acc1, false, false);
        }
        // C/D layout: VGPR r -> row (r + 8*half), col n within tile
#pragma unroll
        for (int r = 0; r < 8; ++r) {
            gatesLds[(r + 8 * half) * GG + 32 * w + n]      = acc0[r];
            gatesLds[(r + 8 * half) * GG + 32 * w + 16 + n] = acc1[r];
        }
        __syncthreads();   // gates complete; Abuf reads done

        // ---- pointwise LSTM cell update (i,f,g,o gate order) ----
#pragma unroll
        for (int r = 0; r < 4; ++r) {
            int idx = tid + 512 * r;
            int row = idx >> 7, hi = idx & 127;
            const float* gr = &gatesLds[row * GG];
            float ig = sigmoidf (gr[hi]            + biasLds[hi]);
            float fg = sigmoidf (gr[HH + hi]       + biasLds[HH + hi]);
            float gg = tanh_fast(gr[2 * HH + hi]   + biasLds[2 * HH + hi]);
            float og = sigmoidf (gr[3 * HH + hi]   + biasLds[3 * HH + hi]);
            float c  = fg * cst[r] + ig * gg;
            cst[r]   = c;
            float h  = og * tanh_fast(c);
            unsigned short hb = f2bf(h);
            Abuf[row * KTOT + KOFF + hi] = hb;           // feeds next step's GEMM
            if constexpr (FIRST) {
                hout[((size_t)(bbase + row) * TT + t) * HH + hi] = hb;
            } else {
                if (t == TT - 1) h2last[(size_t)(bbase + row) * HH + hi] = h;
            }
        }
        // next iteration's __syncthreads() (after input staging) orders h writes vs GEMM
    }
}

// ---------------- FC head: out = relu(h2 @ fc1^T + b1) @ fc2^T + b2 ----------------
__global__ void fc_head(const float* __restrict__ h2, const float* __restrict__ w1,
                        const float* __restrict__ b1, const float* __restrict__ w2,
                        const float* __restrict__ b2, float* __restrict__ out) {
    __shared__ float red[64];
    int b = blockIdx.x, j = threadIdx.x;
    const float* hb = h2 + (size_t)b * HH;
    float s = b1[j];
#pragma unroll 4
    for (int k = 0; k < HH; ++k) s += w1[j * HH + k] * hb[k];
    red[j] = fmaxf(s, 0.f) * w2[j];
    __syncthreads();
    if (j == 0) {
        float a = 0.f;
        for (int k = 0; k < 64; ++k) a += red[k];
        out[b] = a + b2[0];
    }
}

// ---------------- launcher ----------------
extern "C" void kernel_launch(void* const* d_in, const int* in_sizes, int n_in,
                              void* d_out, int out_size, void* d_ws, size_t ws_size,
                              hipStream_t stream) {
    const float* x    = (const float*)d_in[0];
    const float* Wih0 = (const float*)d_in[1];
    const float* Whh0 = (const float*)d_in[2];
    const float* bih0 = (const float*)d_in[3];
    const float* bhh0 = (const float*)d_in[4];
    const float* Wih1 = (const float*)d_in[5];
    const float* Whh1 = (const float*)d_in[6];
    const float* bih1 = (const float*)d_in[7];
    const float* bhh1 = (const float*)d_in[8];
    const float* fc1w = (const float*)d_in[9];
    const float* fc1b = (const float*)d_in[10];
    const float* fc2w = (const float*)d_in[11];
    const float* fc2b = (const float*)d_in[12];
    float* out = (float*)d_out;

    char* ws = (char*)d_ws;
    size_t off = 0;
    unsigned short* Wcat0 = (unsigned short*)(ws + off); off += (size_t)GG * 192 * 2; // 192 KB
    unsigned short* Wcat1 = (unsigned short*)(ws + off); off += (size_t)GG * 256 * 2; // 256 KB
    float* bias0 = (float*)(ws + off); off += GG * 4;
    float* bias1 = (float*)(ws + off); off += GG * 4;
    off = (off + 255) & ~(size_t)255;
    unsigned short* h1 = (unsigned short*)(ws + off); off += (size_t)BB * TT * HH * 2; // 128 MB
    float* h2last = (float*)(ws + off); off += (size_t)BB * HH * 4;

    prep_kernel<<<512, 256, 0, stream>>>(Wih0, Whh0, bih0, bhh0, Wih1, Whh1, bih1, bhh1,
                                         Wcat0, Wcat1, bias0, bias1);
    lstm_pass<192, true ><<<BB / 16, 512, 0, stream>>>(x, nullptr, Wcat0, bias0, h1, nullptr);
    lstm_pass<256, false><<<BB / 16, 512, 0, stream>>>(nullptr, h1, Wcat1, bias1, nullptr, h2last);
    fc_head<<<BB, 64, 0, stream>>>(h2last, fc1w, fc1b, fc2w, fc2b, out);
}
